// DeepSAE_63599875719613
// MI455X (gfx1250) — matrix operand order repaired and checked by reading, not compiled, not run on hardware
//
#include <hip/hip_runtime.h>

typedef __attribute__((ext_vector_type(16))) __bf16 v16bf;
typedef __attribute__((ext_vector_type(8)))  float  v8f;

namespace sae {

constexpr int D_MODEL = 768;
constexpr int D_MID   = 1536;
constexpr int D_FEAT  = 24576;
constexpr int MROWS   = 4 * 512;                      // B*S = 2048
constexpr long long NPRE = (long long)MROWS * D_FEAT; // 50,331,648
constexpr int BM = 128, BN = 128, BK = 32, NTH = 256;

__device__ __forceinline__ unsigned short f2bf(float f) {
  union { float f; unsigned u; } a; a.f = f;
  unsigned r = a.u + 0x7FFFu + ((a.u >> 16) & 1u);   // round-to-nearest-even
  return (unsigned short)(r >> 16);
}

// generic -> LDS(addrspace 3) byte offset (C-style cast performs addrspacecast)
typedef const __attribute__((address_space(3))) void lds_cv;
__device__ __forceinline__ unsigned lds_off(const void* p) {
  return (unsigned)(unsigned long long)(lds_cv*)p;
}

union Frag { v16bf v; uint4 q[2]; };
union U4   { uint4 q; unsigned short s[8]; };

// ---------------- elementwise f32 -> bf16 ----------------
__global__ void cast_kernel(const float* __restrict__ in,
                            unsigned short* __restrict__ out, int n) {
  int i = blockIdx.x * blockDim.x + threadIdx.x;
  if (i < n) out[i] = f2bf(in[i]);
}

// ---------------- [R][C] f32 -> [C][R] bf16 (tiled transpose) ----------------
__global__ void cast_transpose_kernel(const float* __restrict__ in,
                                      unsigned short* __restrict__ out,
                                      int R, int C) {
  __shared__ unsigned short tile[32][33];
  int cT = blockIdx.x * 32, rT = blockIdx.y * 32;
  int tx = threadIdx.x, ty = threadIdx.y;   // blockDim = (32, 8)
#pragma unroll
  for (int j = 0; j < 4; ++j) {
    int r = rT + ty + j * 8;
    tile[ty + j * 8][tx] = f2bf(in[(long long)r * C + cT + tx]);
  }
  __syncthreads();
#pragma unroll
  for (int j = 0; j < 4; ++j) {
    int c = cT + ty + j * 8;
    out[(long long)c * R + rT + tx] = tile[tx][ty + j * 8];
  }
}

// ---------------- WMMA bf16 GEMM: C = act(A[M,K] * Bt[N,K]^T + bias) ----------------
// Block tile 128x128x32, 8 waves (4 in M x 2 in N), each wave 32x64 = 8 WMMA accum.
// Dense path stages tiles with GLOBAL_LOAD_ASYNC_TO_LDS_B128 (ASYNCcnt);
// masked path (top-k) stages through VGPRs so the threshold can be applied.
template<bool RELU, bool MASK, bool FINAL>
__global__ __launch_bounds__(NTH)
void gemm_wmma(const unsigned short* __restrict__ A,    // [M][K] bf16 bits
               const unsigned short* __restrict__ Bt,   // [N][K] bf16 bits (B transposed)
               const float* __restrict__ bias,          // [N]
               unsigned short* __restrict__ Cb,         // bf16 out (!FINAL)
               float* __restrict__ Cf,                  // f32 out (FINAL)
               const float* __restrict__ Xres,          // residual target (FINAL)
               float* __restrict__ lossAcc,             // (FINAL)
               const unsigned* __restrict__ thrPtr,     // bf16-bits threshold (MASK)
               int M, int N, int K) {
  __shared__ alignas(16) unsigned short As[2][BM][BK];
  __shared__ alignas(16) unsigned short Bs[2][BN][BK];

  const int tid  = threadIdx.x;
  const int lane = tid & 31;
  const int wave = tid >> 5;
  const int lh   = lane >> 4;      // lane half (K-split selector per ISA layout)
  const int l16  = lane & 15;
  const int wm   = wave & 3;       // 4 waves in M, 32 rows each
  const int wn   = wave >> 2;      // 2 waves in N, 64 cols each
  const int blockM = blockIdx.y * BM;
  const int blockN = blockIdx.x * BN;

  unsigned thr = 0;
  if constexpr (MASK) thr = *thrPtr;

  v8f acc[2][4];
#pragma unroll
  for (int mi = 0; mi < 2; ++mi)
#pragma unroll
    for (int ni = 0; ni < 4; ++ni) acc[mi][ni] = {};

  // staging: 2 threads per tile-row, each moves 16 halfs (32 B)
  const int srow = tid >> 1;
  const int soff = (tid & 1) << 4;
  const unsigned short* aRow = A  + (long long)(blockM + srow) * K + soff;
  const unsigned short* bRow = Bt + (long long)(blockN + srow) * K + soff;
  const int KT = K / BK;

  // per-lane LDS byte offsets of this thread's staging slots (both buffers)
  unsigned aLds[2], bLds[2];
  aLds[0] = lds_off(&As[0][srow][soff]);
  aLds[1] = lds_off(&As[1][srow][soff]);
  bLds[0] = lds_off(&Bs[0][srow][soff]);
  bLds[1] = lds_off(&Bs[1][srow][soff]);

  auto stage = [&](int kt, int buf) {
    const unsigned short* ap = aRow + (long long)kt * BK;
    const unsigned short* bp = bRow + (long long)kt * BK;
    if (kt + 2 < KT) {                                   // prefetch K+2 tile into L2
      __builtin_prefetch(ap + 2 * BK, 0, 3);
      __builtin_prefetch(bp + 2 * BK, 0, 3);
    }
    if constexpr (MASK) {
      // global top-k mask applied between load and LDS store
      U4 a0, a1, b0, b1;
      a0.q = ((const uint4*)ap)[0];
      a1.q = ((const uint4*)ap)[1];
      b0.q = ((const uint4*)bp)[0];
      b1.q = ((const uint4*)bp)[1];
#pragma unroll
      for (int i = 0; i < 8; ++i) {
        if (a0.s[i] < (unsigned short)thr) a0.s[i] = 0;
        if (a1.s[i] < (unsigned short)thr) a1.s[i] = 0;
      }
      *(uint4*)&As[buf][srow][soff]     = a0.q;
      *(uint4*)&As[buf][srow][soff + 8] = a1.q;
      *(uint4*)&Bs[buf][srow][soff]     = b0.q;
      *(uint4*)&Bs[buf][srow][soff + 8] = b1.q;
    } else {
      // async memory->LDS copy: no VGPR data, tracked by ASYNCcnt.
      // offset: applies to both LDS and global address (mirrored layouts).
      unsigned long long ga = (unsigned long long)ap;
      unsigned long long gb = (unsigned long long)bp;
      unsigned la = aLds[buf];
      unsigned lb = bLds[buf];
      asm volatile("global_load_async_to_lds_b128 %0, %1, off"
                   :: "v"(la), "v"(ga) : "memory");
      asm volatile("global_load_async_to_lds_b128 %0, %1, off offset:16"
                   :: "v"(la), "v"(ga) : "memory");
      asm volatile("global_load_async_to_lds_b128 %0, %1, off"
                   :: "v"(lb), "v"(gb) : "memory");
      asm volatile("global_load_async_to_lds_b128 %0, %1, off offset:16"
                   :: "v"(lb), "v"(gb) : "memory");
    }
  };

  auto waitStage = [&]() {
    if constexpr (!MASK) {
      asm volatile("s_wait_asynccnt 0x0" ::: "memory");
    }
    __syncthreads();
  };

  auto compute = [&](int buf) {
    Frag af[2], bfr[4];
#pragma unroll
    for (int mi = 0; mi < 2; ++mi) {
      // A 16x32 bf16 layout: lane half 0 -> K{0..7,16..23}, half 1 -> K{8..15,24..31}
      const unsigned short* p = &As[buf][wm * 32 + mi * 16 + l16][lh * 8];
      af[mi].q[0] = *(const uint4*)p;
      af[mi].q[1] = *(const uint4*)(p + 16);
    }
#pragma unroll
    for (int ni = 0; ni < 4; ++ni) {
      // B 32x16: lane = column, lane half 0 -> K{0..7,16..23}, half 1 -> K{8..15,24..31}
      const unsigned short* p = &Bs[buf][wn * 64 + ni * 16 + l16][lh * 8];
      bfr[ni].q[0] = *(const uint4*)p;
      bfr[ni].q[1] = *(const uint4*)(p + 16);
    }
#pragma unroll
    for (int mi = 0; mi < 2; ++mi)
#pragma unroll
      for (int ni = 0; ni < 4; ++ni)
        acc[mi][ni] = __builtin_amdgcn_wmma_f32_16x16x32_bf16(
            false, af[mi].v, false, bfr[ni].v, (short)0, acc[mi][ni], false, false);
  };

  int buf = 0;
  stage(0, 0);
  waitStage();
  for (int kt = 0; kt < KT; ++kt) {
    if (kt + 1 < KT) stage(kt + 1, buf ^ 1);
    compute(buf);
    waitStage();
    buf ^= 1;
  }

  // epilogue: C/D layout -> VGPR r: M = r + 8*lanehalf, N = lane%16
  float sq = 0.f;
#pragma unroll
  for (int mi = 0; mi < 2; ++mi) {
    int rbase = blockM + wm * 32 + mi * 16 + lh * 8;
#pragma unroll
    for (int ni = 0; ni < 4; ++ni) {
      int col = blockN + wn * 64 + ni * 16 + l16;
      float bc = bias[col];
#pragma unroll
      for (int r = 0; r < 8; ++r) {
        float v = acc[mi][ni][r] + bc;
        if constexpr (RELU) v = v > 0.f ? v : 0.f;
        long long idx = (long long)(rbase + r) * N + col;
        if constexpr (FINAL) {
          Cf[idx] = v;
          float d = v - Xres[idx];
          sq += d * d;
        } else {
          Cb[idx] = f2bf(v);
        }
      }
    }
  }
  if constexpr (FINAL) {
#pragma unroll
    for (int o = 16; o > 0; o >>= 1) sq += __shfl_down(sq, o, 32);
    if (lane == 0) atomicAdd(lossAcc, sq);
  }
}

// ---------------- top-k threshold: 2-pass byte radix on bf16 bits ----------------
__global__ void hist_hi_kernel(const unsigned short* __restrict__ v,
                               unsigned* __restrict__ hist, long long n8) {
  __shared__ unsigned h[256];
  for (int i = threadIdx.x; i < 256; i += blockDim.x) h[i] = 0;
  __syncthreads();
  long long stride = (long long)gridDim.x * blockDim.x;
  const uint4* v4 = (const uint4*)v;
  for (long long i = (long long)blockIdx.x * blockDim.x + threadIdx.x; i < n8; i += stride) {
    U4 u; u.q = v4[i];
#pragma unroll
    for (int j = 0; j < 8; ++j) atomicAdd(&h[u.s[j] >> 8], 1u);
  }
  __syncthreads();
  for (int i = threadIdx.x; i < 256; i += blockDim.x)
    if (h[i]) atomicAdd(&hist[i], h[i]);
}

__global__ void select_hi_kernel(const unsigned* __restrict__ hist,
                                 const int* __restrict__ kfeat,
                                 long long* __restrict__ sel) {
  long long K = (long long)kfeat[0] * MROWS;
  long long cum = 0;
  int b = 255;
  for (; b >= 0; --b) { cum += hist[b]; if (cum >= K) break; }
  if (b < 0) b = 0;
  long long rank = K - (cum - (long long)hist[b]);
  if (rank < 1) rank = 1;
  sel[0] = b;
  sel[1] = rank;
}

__global__ void hist_lo_kernel(const unsigned short* __restrict__ v,
                               unsigned* __restrict__ hist,
                               const long long* __restrict__ sel, long long n8) {
  __shared__ unsigned h[256];
  for (int i = threadIdx.x; i < 256; i += blockDim.x) h[i] = 0;
  __syncthreads();
  unsigned hi = (unsigned)sel[0];
  long long stride = (long long)gridDim.x * blockDim.x;
  const uint4* v4 = (const uint4*)v;
  for (long long i = (long long)blockIdx.x * blockDim.x + threadIdx.x; i < n8; i += stride) {
    U4 u; u.q = v4[i];
#pragma unroll
    for (int j = 0; j < 8; ++j)
      if ((unsigned)(u.s[j] >> 8) == hi) atomicAdd(&h[u.s[j] & 255u], 1u);
  }
  __syncthreads();
  for (int i = threadIdx.x; i < 256; i += blockDim.x)
    if (h[i]) atomicAdd(&hist[i], h[i]);
}

__global__ void select_lo_kernel(const unsigned* __restrict__ hist,
                                 const long long* __restrict__ sel,
                                 unsigned* __restrict__ thr) {
  long long Kp = sel[1];
  long long cum = 0;
  int l = 255;
  for (; l >= 0; --l) { cum += hist[l]; if (cum >= Kp) break; }
  if (l < 0) l = 0;
  thr[0] = (((unsigned)sel[0]) << 8) | (unsigned)l;
}

__global__ void finalize_kernel(const float* __restrict__ lossAcc,
                                float* __restrict__ out) {
  float l2 = *lossAcc / (float)((long long)MROWS * D_MODEL);
  out[0] = l2;   // loss = l2 + aux(0)
  out[1] = l2;   // l2_loss
}

} // namespace sae

extern "C" void kernel_launch(void* const* d_in, const int* in_sizes, int n_in,
                              void* d_out, int out_size, void* d_ws, size_t ws_size,
                              hipStream_t stream) {
  using namespace sae;
  (void)in_sizes; (void)n_in; (void)out_size; (void)ws_size;

  const float* x      = (const float*)d_in[0];
  const float* W_enc1 = (const float*)d_in[1];   // [768][1536]
  const float* b_enc1 = (const float*)d_in[2];
  const float* W_enc2 = (const float*)d_in[3];   // [1536][24576]
  const float* b_enc2 = (const float*)d_in[4];
  const float* W_dec2 = (const float*)d_in[5];   // [24576][1536]
  const float* b_dec2 = (const float*)d_in[6];
  const float* W_dec1 = (const float*)d_in[7];   // [1536][768]
  const float* b_dec1 = (const float*)d_in[8];
  const int*   k_feat = (const int*)d_in[9];

  char* ws = (char*)d_ws;
  size_t off = 0;
  auto take = [&](size_t bytes) -> char* {
    char* p = ws + off;
    off = (off + bytes + 255) & ~(size_t)255;
    return p;
  };

  unsigned short* xb   = (unsigned short*)take((size_t)MROWS * D_MODEL * 2);
  unsigned short* w1t  = (unsigned short*)take((size_t)D_MID * D_MODEL * 2);   // [1536][768]
  unsigned short* w2t  = (unsigned short*)take((size_t)D_FEAT * D_MID * 2);    // [24576][1536]
  unsigned short* wd2t = (unsigned short*)take((size_t)D_MID * D_FEAT * 2);    // [1536][24576]
  unsigned short* wd1t = (unsigned short*)take((size_t)D_MODEL * D_MID * 2);   // [768][1536]
  unsigned short* mid0 = (unsigned short*)take((size_t)MROWS * D_MID * 2);
  unsigned short* pre1 = (unsigned short*)take((size_t)MROWS * D_FEAT * 2);
  unsigned short* mid2 = (unsigned short*)take((size_t)MROWS * D_MID * 2);
  unsigned*   histHi   = (unsigned*)take(256 * 4);
  unsigned*   histLo   = (unsigned*)take(256 * 4);
  long long*  sel      = (long long*)take(2 * 8);
  unsigned*   thr      = (unsigned*)take(4);
  float*      lossAcc  = (float*)take(4);

  hipMemsetAsync(histHi, 0, 256 * 4, stream);
  hipMemsetAsync(histLo, 0, 256 * 4, stream);
  hipMemsetAsync(lossAcc, 0, 4, stream);

  // ---- one-time-per-launch bf16 casts; weight transposes give [N][K] layout ----
  {
    int n = MROWS * D_MODEL;
    cast_kernel<<<(n + 255) / 256, 256, 0, stream>>>(x, xb, n);
  }
  dim3 tb(32, 8);
  cast_transpose_kernel<<<dim3(D_MID / 32,   D_MODEL / 32), tb, 0, stream>>>(W_enc1, w1t,  D_MODEL, D_MID);
  cast_transpose_kernel<<<dim3(D_FEAT / 32,  D_MID / 32),   tb, 0, stream>>>(W_enc2, w2t,  D_MID,   D_FEAT);
  cast_transpose_kernel<<<dim3(D_MID / 32,   D_FEAT / 32),  tb, 0, stream>>>(W_dec2, wd2t, D_FEAT,  D_MID);
  cast_transpose_kernel<<<dim3(D_MODEL / 32, D_MID / 32),   tb, 0, stream>>>(W_dec1, wd1t, D_MID,   D_MODEL);

  // ---- GEMM1: mid0 = relu(x @ W_enc1 + b_enc1) ----
  gemm_wmma<true, false, false><<<dim3(D_MID / BN, MROWS / BM), NTH, 0, stream>>>(
      xb, w1t, b_enc1, mid0, nullptr, nullptr, nullptr, nullptr, MROWS, D_MID, D_MODEL);

  // ---- GEMM2: pre1 = relu(mid0 @ W_enc2 + b_enc2) ----
  gemm_wmma<true, false, false><<<dim3(D_FEAT / BN, MROWS / BM), NTH, 0, stream>>>(
      mid0, w2t, b_enc2, pre1, nullptr, nullptr, nullptr, nullptr, MROWS, D_FEAT, D_MID);

  // ---- global top-k threshold (exact on bf16 bits; relu => nonneg => monotonic) ----
  hist_hi_kernel<<<2048, 256, 0, stream>>>(pre1, histHi, NPRE / 8);
  select_hi_kernel<<<1, 1, 0, stream>>>(histHi, k_feat, sel);
  hist_lo_kernel<<<2048, 256, 0, stream>>>(pre1, histLo, sel, NPRE / 8);
  select_lo_kernel<<<1, 1, 0, stream>>>(histLo, sel, thr);

  // ---- GEMM3: mid2 = relu(mask(pre1) @ W_dec2 + b_dec2), mask applied in staging ----
  gemm_wmma<true, true, false><<<dim3(D_MID / BN, MROWS / BM), NTH, 0, stream>>>(
      pre1, wd2t, b_dec2, mid2, nullptr, nullptr, nullptr, thr, MROWS, D_MID, D_FEAT);

  // ---- GEMM4: recon = mid2 @ W_dec1 + b_dec1 (fp32 out) + MSE accumulation ----
  float* recon = (float*)d_out;
  gemm_wmma<false, false, true><<<dim3(D_MODEL / BN, MROWS / BM), NTH, 0, stream>>>(
      mid2, wd1t, b_dec1, nullptr, recon, x, lossAcc, nullptr, MROWS, D_MODEL, D_MID);

  finalize_kernel<<<1, 1, 0, stream>>>(lossAcc, recon + (size_t)MROWS * D_MODEL);
}
